// GAE_Encoder_30700426232142
// MI455X (gfx1250) — compile-verified, matrix operand order checked by reading, not run on hardware
//
#include <hip/hip_runtime.h>

// GCN 2-layer forward for MI455X (gfx1250), wave32.
// GEMMs: v_wmma_f32_16x16x32_f16 with pre-swizzled f16 weights (B in exact
// per-lane WMMA layout -> contiguous b128 loads) and f16 activations.
// Edge aggregation: coalesced gather + hardware global_atomic_add_f32 scatter.

typedef __attribute__((ext_vector_type(16))) _Float16 v16h;
typedef __attribute__((ext_vector_type(8)))  _Float16 v8h;
typedef __attribute__((ext_vector_type(8)))  float    v8f;

#define CH_IN  128
#define CH_HID 128
#define CH_OUT 64

// ---------------- degree / normalization ----------------

__global__ void k_fill1(float* __restrict__ p, int n) {
  int i = blockIdx.x * blockDim.x + threadIdx.x;
  if (i < n) p[i] = 1.0f;  // self-loop contributes 1 to every degree
}

__global__ void k_deg_accum(const int* __restrict__ col, float* __restrict__ deg, int E) {
  int e = blockIdx.x * blockDim.x + threadIdx.x;
  if (e < E)
    __hip_atomic_fetch_add(&deg[col[e]], 1.0f, __ATOMIC_RELAXED, __HIP_MEMORY_SCOPE_AGENT);
}

__global__ void k_rsqrt(float* __restrict__ p, int n) {
  int i = blockIdx.x * blockDim.x + threadIdx.x;
  if (i < n) p[i] = rsqrtf(p[i]);  // deg >= 1 always (self-loops)
}

// ---------------- f32 -> f16 elementwise ----------------

__global__ void k_cvt_f16(const float* __restrict__ src, _Float16* __restrict__ dst,
                          long long n) {
  long long i = (long long)blockIdx.x * blockDim.x + threadIdx.x;
  if (i < n) dst[i] = (_Float16)src[i];
}

// ---------------- weight pre-swizzle into WMMA B layout ----------------
// Bs[ntile][ktile][lane][j] (16 halves per lane, contiguous 32B) where
//   lane: idx16 = lane&15 -> column n0+idx16; khalf = (lane>>4)*8
//   j<8 : K = ktile*32 + khalf + j ; j>=8 : K = ktile*32 + khalf + 16 + (j-8)
template<int K, int NOUT>
__global__ void k_swizzleB(const float* __restrict__ B, _Float16* __restrict__ Bs) {
  const int total = (NOUT / 16) * (K / 32) * 512;
  int i = blockIdx.x * blockDim.x + threadIdx.x;
  if (i >= total) return;
  int j     = i & 15;
  int lane  = (i >> 4) & 31;
  int tile  = i >> 9;
  int ktile = tile % (K / 32);
  int ntile = tile / (K / 32);
  int idx16 = lane & 15;
  int khalf = (lane >> 4) * 8;
  int k = ktile * 32 + khalf + (j & 7) + ((j >> 3) * 16);
  int n = ntile * 16 + idx16;
  Bs[i] = (_Float16)B[(size_t)k * NOUT + n];
}

// ---------------- dense GEMM via WMMA ----------------
// C[M x NOUT] = A[M x K] @ B[K x NOUT]; one wave -> 16x16 tile, f32 accum.
// A: row-major f16. Bs: pre-swizzled f16. blockDim = (NOUT/16)*32, grid = M/16.
template<int K, int NOUT>
__global__ void k_gemm_wmma(const _Float16* __restrict__ A,
                            const _Float16* __restrict__ Bs,
                            float* __restrict__ C) {
  const int wave  = threadIdx.x >> 5;
  const int lane  = threadIdx.x & 31;
  const size_t m0 = (size_t)blockIdx.x * 16;
  const int idx16 = lane & 15;
  const int khalf = (lane >> 4) * 8;
  const _Float16* arow = A + (m0 + idx16) * K + khalf;
  const _Float16* brow = Bs + (size_t)wave * (K / 32) * 512 + lane * 16;
  v8f acc = {};
#pragma unroll
  for (int kt = 0; kt < K / 32; ++kt) {
    v8h lo = *(const v8h*)(arow + kt * 32);        // K = kt*32+khalf+0..7
    v8h hi = *(const v8h*)(arow + kt * 32 + 16);   // K = kt*32+khalf+16..23
    v16h a;
#pragma unroll
    for (int j = 0; j < 8; ++j) { a[j] = lo[j]; a[j + 8] = hi[j]; }
    v16h b = *(const v16h*)(brow + (size_t)kt * 512);
    acc = __builtin_amdgcn_wmma_f32_16x16x32_f16(
        /*neg_a=*/false, a, /*neg_b=*/false, b,
        /*c_mod=*/(short)0, acc, /*reuse_a=*/false, /*reuse_b=*/false);
  }
  // C/D layout: VGPR r -> M = r + 8*(lane>>4), N = lane&15
  const int mbase = (lane >> 4) * 8;
  const int n0 = wave * 16;
  float* crow = C + (m0 + mbase) * NOUT + (n0 + idx16);
#pragma unroll
  for (int r = 0; r < 8; ++r) crow[(size_t)r * NOUT] = acc[r];
}

// ---------------- self-loop init (+ optional bias) ----------------
// out[i,c] = xw[i,c] * dinv[i]^2 (+ bias[c]); fully overwrites dst.
template<int F, bool BIAS>
__global__ void k_selfloop(const float* __restrict__ xw, const float* __restrict__ dinv,
                           const float* __restrict__ bias, float* __restrict__ out,
                           long long total) {
  long long i = (long long)blockIdx.x * blockDim.x + threadIdx.x;
  if (i >= total) return;
  int node = (int)(i / F);
  int c    = (int)(i % F);
  float di = dinv[node];
  float v  = xw[i] * di * di;
  if (BIAS) v += bias[c];
  out[i] = v;
}

// ---------------- edge scatter-add ----------------
// One wave per edge; each lane handles F/32 contiguous channels.
template<int F>
__global__ void k_agg(const float* __restrict__ src, const int* __restrict__ row,
                      const int* __restrict__ col, const float* __restrict__ dinv,
                      float* __restrict__ dst, int E) {
  const int e    = blockIdx.x * 8 + (threadIdx.x >> 5);  // blockDim = 256 = 8 waves
  const int lane = threadIdx.x & 31;
  if (e >= E) return;
  const int r = row[e];
  const int c = col[e];
  const float nrm = dinv[r] * dinv[c];
  constexpr int V = F / 32;
  const float* s = src + (size_t)r * F + lane * V;
  float*       d = dst + (size_t)c * F + lane * V;
  float vals[V];
#pragma unroll
  for (int j = 0; j < V; ++j) vals[j] = s[j];
#pragma unroll
  for (int j = 0; j < V; ++j)
    __hip_atomic_fetch_add(d + j, vals[j] * nrm, __ATOMIC_RELAXED, __HIP_MEMORY_SCOPE_AGENT);
}

// ---------------- bias + relu, emit f16 activations for next GEMM ----------------

__global__ void k_bias_relu_f16(const float* __restrict__ h, const float* __restrict__ b,
                                _Float16* __restrict__ out, long long total) {
  long long i = (long long)blockIdx.x * blockDim.x + threadIdx.x;
  if (i >= total) return;
  float v = h[i] + b[(int)(i % CH_HID)];
  out[i] = (_Float16)(v > 0.0f ? v : 0.0f);
}

// ---------------- launch ----------------

extern "C" void kernel_launch(void* const* d_in, const int* in_sizes, int n_in,
                              void* d_out, int out_size, void* d_ws, size_t ws_size,
                              hipStream_t stream) {
  const float* x   = (const float*)d_in[0];
  const int*   ei  = (const int*)d_in[1];
  const float* W1  = (const float*)d_in[2];
  const float* b1  = (const float*)d_in[3];
  const float* W2  = (const float*)d_in[4];
  const float* b2  = (const float*)d_in[5];
  float*       out = (float*)d_out;

  const int N = in_sizes[0] / CH_IN;   // 100000 (divisible by 16)
  const int E = in_sizes[1] / 2;       // 1600000
  const int* row = ei;                 // edge_index[0] = sources
  const int* col = ei + E;             // edge_index[1] = targets

  // workspace: dinv[N] | xw f32[N*128] | h1 f32[N*128] | ah f16[N*128] | W1s | W2s
  float* dinv = (float*)d_ws;
  size_t npad = ((size_t)N + 1023) & ~(size_t)1023;
  float* xw = dinv + npad;
  float* h1 = xw + (size_t)N * CH_HID;
  _Float16* ah  = (_Float16*)(h1 + (size_t)N * CH_HID);  // x(f16), later relu(h1)(f16)
  _Float16* W1s = ah + (size_t)N * CH_HID;
  _Float16* W2s = W1s + CH_IN * CH_HID;

  const int T = 256;
  // normalization: deg = 1 + indegree; dinv = rsqrt(deg)
  k_fill1<<<(N + T - 1) / T, T, 0, stream>>>(dinv, N);
  k_deg_accum<<<(E + T - 1) / T, T, 0, stream>>>(col, dinv, E);
  k_rsqrt<<<(N + T - 1) / T, T, 0, stream>>>(dinv, N);

  // weight swizzle + x -> f16
  {
    const int t1 = (CH_HID / 16) * (CH_IN / 32) * 512;
    const int t2 = (CH_OUT / 16) * (CH_HID / 32) * 512;
    k_swizzleB<CH_IN, CH_HID><<<(t1 + T - 1) / T, T, 0, stream>>>(W1, W1s);
    k_swizzleB<CH_HID, CH_OUT><<<(t2 + T - 1) / T, T, 0, stream>>>(W2, W2s);
    long long tot = (long long)N * CH_IN;
    k_cvt_f16<<<(int)((tot + T - 1) / T), T, 0, stream>>>(x, ah, tot);
  }

  // ---- layer 1: xw = x @ W1 ; h1 = scatter(norm * xw) ; relu(+b1) -> f16 ----
  k_gemm_wmma<CH_IN, CH_HID><<<N / 16, (CH_HID / 16) * 32, 0, stream>>>(ah, W1s, xw);
  {
    long long tot = (long long)N * CH_HID;
    k_selfloop<CH_HID, false><<<(int)((tot + T - 1) / T), T, 0, stream>>>(
        xw, dinv, nullptr, h1, tot);
    k_agg<CH_HID><<<(E + 7) / 8, T, 0, stream>>>(xw, row, col, dinv, h1, E);
    k_bias_relu_f16<<<(int)((tot + T - 1) / T), T, 0, stream>>>(h1, b1, ah, tot);
  }

  // ---- layer 2: xw2 = h1 @ W2 ; out = scatter(norm * xw2) + b2 ----
  k_gemm_wmma<CH_HID, CH_OUT><<<N / 16, (CH_OUT / 16) * 32, 0, stream>>>(ah, W2s, xw);
  {
    long long tot = (long long)N * CH_OUT;
    k_selfloop<CH_OUT, true><<<(int)((tot + T - 1) / T), T, 0, stream>>>(
        xw, dinv, b2, out, tot);
    k_agg<CH_OUT><<<(E + 7) / 8, T, 0, stream>>>(xw, row, col, dinv, out, E);
  }
}
